// RON_40948218200259
// MI455X (gfx1250) — compile-verified, matrix operand order checked by reading, not run on hardware
//
#include <hip/hip_runtime.h>
#include <hip/hip_bf16.h>
#include <math.h>

typedef __attribute__((ext_vector_type(16))) __bf16 v16bf;
typedef __attribute__((ext_vector_type(8)))  __bf16 v8bf;
typedef __attribute__((ext_vector_type(8)))  float  v8f;
typedef __attribute__((ext_vector_type(4)))  float  v4f;

#define RON_DT 0.042f
#define RON_B  64
#define RON_L  1024
#define RON_I  128
#define RON_H  512

// Fragment element count for h2h: 32 ntiles * 16 ktiles * 32 lanes * 16 elems
#define H2H_FRAG_ELEMS (32 * 16 * 32 * 16)   // 262144
// x2h: 32 ntiles * 4 ktiles * 32 lanes * 16 elems
#define X2H_FRAG_ELEMS (32 * 4 * 32 * 16)    // 65536

// ---------------------------------------------------------------------------
// Pre-swizzle h2h (512x512) and x2h (128x512) f32 -> bf16 into the exact
// per-lane B-fragment layout of v_wmma_f32_16x16x32_bf16, so the hot loops
// can fetch a whole lane fragment with two contiguous global_load_b128.
// Layout: frag[((nt*KT + kt)*32 + lane)*16 + j], j = 2*v + p:
//   n = nt*16 + (lane&15); khalf = lane>>4
//   k = kt*32 + khalf*8 + (v<4 ? 2v : 16 + 2(v-4)) + p
// ---------------------------------------------------------------------------
__global__ __launch_bounds__(256)
void RON_prep_frags(const float* __restrict__ h2h, const float* __restrict__ x2h,
                    __bf16* __restrict__ h2h_frag, __bf16* __restrict__ x2h_frag) {
  int id = blockIdx.x * blockDim.x + threadIdx.x;
  if (id < H2H_FRAG_ELEMS) {
    int j    = id & 15;
    int lane = (id >> 4) & 31;
    int kt   = (id >> 9) & 15;
    int nt   = id >> 13;
    int n     = nt * 16 + (lane & 15);
    int khalf = lane >> 4;
    int v = j >> 1, p = j & 1;
    int k = kt * 32 + khalf * 8 + ((v < 4) ? (2 * v) : (16 + 2 * (v - 4))) + p;
    h2h_frag[id] = (__bf16)h2h[(size_t)k * RON_H + n];
  } else if (id < H2H_FRAG_ELEMS + X2H_FRAG_ELEMS) {
    int id2  = id - H2H_FRAG_ELEMS;
    int j    = id2 & 15;
    int lane = (id2 >> 4) & 31;
    int kt   = (id2 >> 9) & 3;
    int nt   = id2 >> 11;
    int n     = nt * 16 + (lane & 15);
    int khalf = lane >> 4;
    int v = j >> 1, p = j & 1;
    int k = kt * 32 + khalf * 8 + ((v < 4) ? (2 * v) : (16 + 2 * (v - 4))) + p;
    x2h_frag[id2] = (__bf16)x2h[(size_t)k * RON_H + n];
  }
}

// ---------------------------------------------------------------------------
// Input projection: u = x @ x2h + bias, staged into d_out's all_states region
// (overwritten in place by the scan). M = B*L = 65536, N = 512, K = 128.
// Block = 256 threads (8 waves); block computes 16 rows x 512 cols;
// wave w covers n-tiles [4w, 4w+4). Grid = 4096 blocks.
// ---------------------------------------------------------------------------
__global__ __launch_bounds__(256)
void RON_proj(const float* __restrict__ x, const __bf16* __restrict__ x2h_frag,
              const float* __restrict__ bias, float* __restrict__ u) {
  const int lane  = threadIdx.x & 31;
  const int wave  = threadIdx.x >> 5;     // 0..7
  const int row0  = blockIdx.x * 16;      // M base (flattened b*L + t)
  const int mrow  = row0 + (lane & 15);
  const int khalf = lane >> 4;

  v8f acc[4] = {};

#pragma unroll
  for (int kt = 0; kt < 4; ++kt) {
    // A fragment: x[mrow, kt*32 + khalf*8 + {0..7}] and +16 .. +23 (f32 -> bf16)
    const float* xp = x + (size_t)mrow * RON_I + kt * 32 + khalf * 8;
    v4f a0 = *(const v4f*)(xp);
    v4f a1 = *(const v4f*)(xp + 4);
    v4f a2 = *(const v4f*)(xp + 16);
    v4f a3 = *(const v4f*)(xp + 20);
    v16bf a;
#pragma unroll
    for (int i = 0; i < 4; ++i) {
      a[i]      = (__bf16)a0[i];
      a[4 + i]  = (__bf16)a1[i];
      a[8 + i]  = (__bf16)a2[i];
      a[12 + i] = (__bf16)a3[i];
    }
#pragma unroll
    for (int nt = 0; nt < 4; ++nt) {
      int ntg = wave * 4 + nt;
      const v16bf b = *(const v16bf*)(x2h_frag + (((size_t)ntg * 4 + kt) * 32 + lane) * 16);
      acc[nt] = __builtin_amdgcn_wmma_f32_16x16x32_bf16(
          false, a, false, b, (short)0, acc[nt], false, false);
    }
  }

  // C layout: lane<16 -> (m = row0+v, n = base+lane); lane>=16 -> (m = row0+v+8)
#pragma unroll
  for (int nt = 0; nt < 4; ++nt) {
    int n = (wave * 4 + nt) * 16 + (lane & 15);
    float bi = bias[n];
#pragma unroll
    for (int v = 0; v < 8; ++v) {
      int m = row0 + v + khalf * 8;
      u[(size_t)m * RON_H + n] = acc[nt][v] + bi;
    }
  }
}

// ---------------------------------------------------------------------------
// Recurrent scan. 4 blocks x 512 threads (16 waves). Block handles batch rows
// [b0, b0+16). Wave w owns n-tiles {2w, 2w+1} (columns [32w, 32w+32)).
// hy/hz live in f32 registers in C-matrix layout; bf16 copy of hy is shared
// through LDS (row stride 520 to avoid bank conflicts) as the WMMA A operand.
// u is read from d_out and overwritten in place with hy_t.
// ---------------------------------------------------------------------------
__global__ __launch_bounds__(512)
void RON_scan(const __bf16* __restrict__ h2h_frag,
              const float* __restrict__ gamma, const float* __restrict__ eps,
              float* __restrict__ out) {
  __shared__ __bf16 hyb[16 * 520];

  const int lane  = threadIdx.x & 31;
  const int wave  = threadIdx.x >> 5;   // 0..15
  const int b0    = blockIdx.x * 16;
  const int khalf = lane >> 4;
  const int nl    = lane & 15;

  float hy[2][8], hz[2][8];
  float g[2], e[2];
#pragma unroll
  for (int q = 0; q < 2; ++q) {
    int n = (wave * 2 + q) * 16 + nl;
    g[q] = gamma[n];
    e[q] = eps[n];
#pragma unroll
    for (int v = 0; v < 8; ++v) { hy[q][v] = 0.f; hz[q][v] = 0.f; }
  }
  for (int i = threadIdx.x; i < 16 * 520; i += 512) hyb[i] = (__bf16)0.f;
  __syncthreads();

  const size_t fin_base = (size_t)RON_B * RON_L * RON_H;

  for (int t = 0; t < RON_L; ++t) {
    v8f acc0 = {}, acc1 = {};
    // W = hy @ h2h for this wave's 32 columns: 16 k-tiles of 32
#pragma unroll 4
    for (int kt = 0; kt < 16; ++kt) {
      // A fragment from LDS: lane<16 -> row nl, K kt*32+{0..7,16..23};
      // lane>=16 -> row nl, K kt*32+{8..15,24..31} (khalf*8 shift)
      const __bf16* ap = hyb + ((size_t)nl * 520 + kt * 32 + khalf * 8);
      v8bf c0 = *(const v8bf*)(ap);
      v8bf c1 = *(const v8bf*)(ap + 16);
      v16bf a;
#pragma unroll
      for (int i = 0; i < 8; ++i) { a[i] = c0[i]; a[8 + i] = c1[i]; }

      const v16bf bf0 = *(const v16bf*)(h2h_frag + (((size_t)(wave * 2 + 0) * 16 + kt) * 32 + lane) * 16);
      const v16bf bf1 = *(const v16bf*)(h2h_frag + (((size_t)(wave * 2 + 1) * 16 + kt) * 32 + lane) * 16);
      acc0 = __builtin_amdgcn_wmma_f32_16x16x32_bf16(false, a, false, bf0, (short)0, acc0, false, false);
      acc1 = __builtin_amdgcn_wmma_f32_16x16x32_bf16(false, a, false, bf1, (short)0, acc1, false, false);
    }
    __syncthreads();  // all waves done reading hyb for step t

    // Elementwise update on the accumulator-owning lanes
#pragma unroll
    for (int q = 0; q < 2; ++q) {
      int n = (wave * 2 + q) * 16 + nl;
#pragma unroll
      for (int v = 0; v < 8; ++v) {
        int m = v + khalf * 8;
        size_t uidx = ((size_t)(b0 + m) * RON_L + t) * RON_H + n;
        float w = (q == 0) ? acc0[v] : acc1[v];
        float uval = out[uidx];                 // staged u = x@x2h + bias
        float z = tanhf(uval + w);
        float hyv = hy[q][v];
        float hzv = hz[q][v];
        hzv = hzv + RON_DT * (z - g[q] * hyv - e[q] * hzv);
        hyv = hyv + RON_DT * hzv;
        hz[q][v] = hzv;
        hy[q][v] = hyv;
        out[uidx] = hyv;                        // all_states[b, t, n] (overwrites u)
        hyb[m * 520 + n] = (__bf16)hyv;         // A operand for step t+1
        if (t == RON_L - 1)
          out[fin_base + (size_t)(b0 + m) * RON_H + n] = hyv;  // hy_final
      }
    }
    __syncthreads();  // hyb fully rewritten before step t+1 reads
  }
}

// ---------------------------------------------------------------------------
extern "C" void kernel_launch(void* const* d_in, const int* in_sizes, int n_in,
                              void* d_out, int out_size, void* d_ws, size_t ws_size,
                              hipStream_t stream) {
  (void)in_sizes; (void)n_in; (void)out_size; (void)ws_size;
  const float* x     = (const float*)d_in[0];
  const float* x2h   = (const float*)d_in[1];
  const float* h2h   = (const float*)d_in[2];
  const float* bias  = (const float*)d_in[3];
  const float* gamma = (const float*)d_in[4];
  const float* eps   = (const float*)d_in[5];
  float* out = (float*)d_out;

  __bf16* h2h_frag = (__bf16*)d_ws;
  __bf16* x2h_frag = h2h_frag + H2H_FRAG_ELEMS;

  // 1) swizzle weights into WMMA fragment order (bf16)
  RON_prep_frags<<<(H2H_FRAG_ELEMS + X2H_FRAG_ELEMS + 255) / 256, 256, 0, stream>>>(
      h2h, x2h, h2h_frag, x2h_frag);

  // 2) u = x @ x2h + bias, staged into d_out (overwritten in place by scan)
  RON_proj<<<(RON_B * RON_L) / 16, 256, 0, stream>>>(x, x2h_frag, bias, out);

  // 3) serial-in-L recurrent scan, 4 workgroups x 16 batch rows
  RON_scan<<<RON_B / 16, 512, 0, stream>>>(h2h_frag, gamma, eps, out);
}
